// MultiReDiffusion_63788854280903
// MI455X (gfx1250) — compile-verified, matrix operand order checked by reading, not run on hardware
//
#include <hip/hip_runtime.h>
#include <hip/hip_bf16.h>

typedef __attribute__((ext_vector_type(16))) __bf16 v16bf;
typedef __attribute__((ext_vector_type(8)))  __bf16 v8bf;
typedef __attribute__((ext_vector_type(8)))  float  v8f;
typedef __attribute__((ext_vector_type(4)))  float  v4f;

#define R_     4
#define S_     8
#define N_     4096
#define DIN    128
#define DOUT   128
#define TILE_M 64
#define KSTAGE 64       // k-chunk staged to LDS per barrier pair
#define XT_STRIDE 72    // 64 k + 8 pad (bf16) : 144B rows -> conflict-free, 16B aligned
#define DF_STRIDE 136   // 128 d + 8 pad (bf16): 272B rows -> conflict-free, 16B aligned

static __device__ __forceinline__ v16bf cat8(v8bf lo, v8bf hi) {
    return __builtin_shufflevector(lo, hi, 0,1,2,3,4,5,6,7,8,9,10,11,12,13,14,15);
}

static __device__ __forceinline__ v16bf cvt16(v4f a0, v4f a1, v4f a2, v4f a3) {
    v16bf f;
    #pragma unroll
    for (int j = 0; j < 4; ++j) {
        f[j]      = (__bf16)a0[j];
        f[4 + j]  = (__bf16)a1[j];
        f[8 + j]  = (__bf16)a2[j];
        f[12 + j] = (__bf16)a3[j];
    }
    return f;
}

__global__ __launch_bounds__(512, 1)
void multirediff_fused(const float* __restrict__ gamma,
                       const float* __restrict__ Tm,
                       const float* __restrict__ a,
                       const float* __restrict__ x,
                       const float* __restrict__ W,
                       const float* __restrict__ bias,
                       const float* __restrict__ mix_w,
                       const float* __restrict__ mix_b,
                       const float* __restrict__ alpha,
                       float* __restrict__ out)
{
    __shared__ __bf16 xT[DIN * XT_STRIDE];       // x chunk, transposed, bf16
    __shared__ __bf16 diffs[TILE_M * DF_STRIDE]; // one relation's diff tile, bf16

    const int tid  = threadIdx.x;
    const int lane = tid & 31;
    const int wave = tid >> 5;
    const int lg   = lane >> 4;   // 0: K 0..7 & 16..23, 1: K 8..15 & 24..31
    const int ln   = lane & 15;
    const int ws   = wave & 3;    // row strip (16 rows)
    const int cq   = wave >> 2;   // column quarter (32 cols)
    const int row0 = blockIdx.x * TILE_M;

    // ---- coef[r] = sum_s softmax(gamma[r])_s * T[r,s]  (tiny, redundant per thread)
    float coef[R_];
    #pragma unroll
    for (int r = 0; r < R_; ++r) {
        float g[S_], mx = -3.4e38f;
        #pragma unroll
        for (int s = 0; s < S_; ++s) { g[s] = gamma[r * S_ + s]; mx = fmaxf(mx, g[s]); }
        float se = 0.f, st = 0.f;
        #pragma unroll
        for (int s = 0; s < S_; ++s) {
            float e = __expf(g[s] - mx);
            se += e;
            st += e * Tm[r * S_ + s];
        }
        coef[r] = st / se;
    }
    float mw[R_ * R_];
    #pragma unroll
    for (int i = 0; i < R_ * R_; ++i) mw[i] = mix_w[i];
    float mb[R_];
    #pragma unroll
    for (int q = 0; q < R_; ++q) mb[q] = mix_b[q];
    const float al = alpha[0];

    // per-relation row pointers into a
    const float* __restrict__ arow[R_];
    #pragma unroll
    for (int r = 0; r < R_; ++r)
        arow[r] = a + (size_t)r * N_ * N_ + (size_t)(row0 + 16 * ws + ln) * N_;

    // GEMM1 accumulators for all 4 relations: 4 * 2 * v8f = 64 VGPRs
    v8f acc[R_][2];
    #pragma unroll
    for (int r = 0; r < R_; ++r)
        #pragma unroll
        for (int t = 0; t < 2; ++t)
            #pragma unroll
            for (int i = 0; i < 8; ++i) acc[r][t][i] = 0.f;

    // mixed[q] accumulators: 64 VGPRs
    v8f mixed[R_][2];
    #pragma unroll
    for (int q = 0; q < R_; ++q)
        #pragma unroll
        for (int t = 0; t < 2; ++t)
            #pragma unroll
            for (int i = 0; i < 8; ++i) mixed[q][t][i] = 0.f;

    // =========== GEMM1: diff[r] = a[r](64 x 4096) @ x(4096 x 128), k-outer / r-inner
    for (int k0 = 0; k0 < N_; k0 += KSTAGE) {
        __syncthreads();  // previous xT consumption complete
        // stage x[k0:k0+64][0:128] -> xT transposed bf16 (512 thr * 16 elems)
        {
            const int k  = tid >> 3;          // 0..63
            const int n0 = (tid & 7) * 16;    // 0..112
            const float* xp = x + (size_t)(k0 + k) * DIN + n0;
            v4f x0 = *(const v4f*)xp;
            v4f x1 = *(const v4f*)(xp + 4);
            v4f x2 = *(const v4f*)(xp + 8);
            v4f x3 = *(const v4f*)(xp + 12);
            #pragma unroll
            for (int j = 0; j < 4; ++j) {
                xT[(n0 + j)      * XT_STRIDE + k] = (__bf16)x0[j];
                xT[(n0 + 4 + j)  * XT_STRIDE + k] = (__bf16)x1[j];
                xT[(n0 + 8 + j)  * XT_STRIDE + k] = (__bf16)x2[j];
                xT[(n0 + 12 + j) * XT_STRIDE + k] = (__bf16)x3[j];
            }
        }
        __syncthreads();

        #pragma unroll
        for (int sub = 0; sub < 2; ++sub) {
            const int kk = k0 + 32 * sub;
            // B fragments once per substep, shared by all 4 relations
            v16bf bfrag[2];
            #pragma unroll
            for (int t = 0; t < 2; ++t) {
                const int n = 32 * cq + 16 * t + ln;
                const __bf16* xp0 = &xT[n * XT_STRIDE + 32 * sub + 8 * lg];
                v8bf blo = *(const v8bf*)xp0;         // K 8lg..8lg+8
                v8bf bhi = *(const v8bf*)(xp0 + 16);  // K 16+8lg..
                bfrag[t] = cat8(blo, bhi);
            }
            #pragma unroll
            for (int r = 0; r < R_; ++r) {
                const float* ap = arow[r] + kk + 8 * lg;
                v16bf afrag = cvt16(*(const v4f*)ap,
                                    *(const v4f*)(ap + 4),
                                    *(const v4f*)(ap + 16),
                                    *(const v4f*)(ap + 20));
                #pragma unroll
                for (int t = 0; t < 2; ++t)
                    acc[r][t] = __builtin_amdgcn_wmma_f32_16x16x32_bf16(
                        false, afrag, false, bfrag[t], (short)0, acc[r][t], false, false);
            }
        }
        // prefetch next chunk of the a stream (2 cachelines per row)
        #pragma unroll
        for (int r = 0; r < R_; ++r) {
            __builtin_prefetch(arow[r] + k0 + KSTAGE, 0, 1);
            __builtin_prefetch(arow[r] + k0 + KSTAGE + 32, 0, 1);
        }
    }

    // =========== epilogue per relation: scale -> LDS -> GEMM2 vs W[r]^T -> mix
    for (int r = 0; r < R_; ++r) {
        __syncthreads();  // previous diffs reads (or xT reads) complete
        {
            const float cr = coef[r];
            #pragma unroll
            for (int t = 0; t < 2; ++t)
                #pragma unroll
                for (int i = 0; i < 8; ++i) {
                    const int m = 16 * ws + i + 8 * lg;
                    const int c = 32 * cq + 16 * t + ln;
                    diffs[m * DF_STRIDE + c] = (__bf16)(acc[r][t][i] * cr);
                }
        }
        __syncthreads();

        // lin = diff(64x128) @ W[r]^T(128x128) + b[r]; bias folded into C
        v8f lacc[2];
        #pragma unroll
        for (int t = 0; t < 2; ++t) {
            const float bv = bias[r * DOUT + 32 * cq + 16 * t + ln];
            #pragma unroll
            for (int i = 0; i < 8; ++i) lacc[t][i] = bv;
        }
        const float* __restrict__ Wr = W + (size_t)r * DOUT * DIN;
        #pragma unroll
        for (int d0 = 0; d0 < DIN; d0 += 32) {
            const __bf16* dp = &diffs[(16 * ws + ln) * DF_STRIDE + d0 + 8 * lg];
            v8bf alo = *(const v8bf*)dp;
            v8bf ahi = *(const v8bf*)(dp + 16);
            v16bf afrag2 = cat8(alo, ahi);
            #pragma unroll
            for (int t = 0; t < 2; ++t) {
                const int o = 32 * cq + 16 * t + ln;
                const float* wp = Wr + (size_t)o * DIN + d0 + 8 * lg;
                v16bf bfrag2 = cvt16(*(const v4f*)wp,
                                     *(const v4f*)(wp + 4),
                                     *(const v4f*)(wp + 16),
                                     *(const v4f*)(wp + 20));
                lacc[t] = __builtin_amdgcn_wmma_f32_16x16x32_bf16(
                    false, afrag2, false, bfrag2, (short)0, lacc[t], false, false);
            }
        }

        // mixed[q] += mix_w[q,r] * lin
        #pragma unroll
        for (int q = 0; q < R_; ++q) {
            const float w = mw[q * R_ + r];
            #pragma unroll
            for (int t = 0; t < 2; ++t)
                #pragma unroll
                for (int i = 0; i < 8; ++i)
                    mixed[q][t][i] += w * lacc[t][i];
        }
    }

    // =========== PReLU + sum over q, write f32 output
    #pragma unroll
    for (int t = 0; t < 2; ++t)
        #pragma unroll
        for (int i = 0; i < 8; ++i) {
            float s = 0.f;
            #pragma unroll
            for (int q = 0; q < R_; ++q) {
                const float v = mixed[q][t][i] + mb[q];
                s += (v >= 0.f) ? v : al * v;
            }
            const int m = row0 + 16 * ws + i + 8 * lg;
            const int c = 32 * cq + 16 * t + ln;
            out[(size_t)m * DOUT + c] = s;
        }
}

extern "C" void kernel_launch(void* const* d_in, const int* in_sizes, int n_in,
                              void* d_out, int out_size, void* d_ws, size_t ws_size,
                              hipStream_t stream) {
    const float* gamma = (const float*)d_in[0];
    const float* Tm    = (const float*)d_in[1];
    const float* a     = (const float*)d_in[2];
    const float* x     = (const float*)d_in[3];
    const float* W     = (const float*)d_in[4];
    const float* bias  = (const float*)d_in[5];
    const float* mixw  = (const float*)d_in[6];
    const float* mixb  = (const float*)d_in[7];
    const float* alpha = (const float*)d_in[8];
    float* out = (float*)d_out;

    dim3 grid(N_ / TILE_M);   // 64 blocks
    dim3 block(512);          // 16 wave32 waves
    hipLaunchKernelGGL(multirediff_fused, grid, block, 0, stream,
                       gamma, Tm, a, x, W, bias, mixw, mixb, alpha, out);
}
